// DeformableStripConv_28381143892652
// MI455X (gfx1250) — matrix operand order repaired and checked by reading, not compiled, not run on hardware
//
#include <hip/hip_runtime.h>

// ---------------------------------------------------------------------------
// Problem constants
// ---------------------------------------------------------------------------
#define BB    8
#define CIN   128
#define COUT  128
#define HH    96
#define WWW   96
#define HWPIX (HH * WWW)          // 9216
#define NPIX  (BB * HWPIX)        // 73728
#define KKT   9                   // 3x3 taps
#define KTOT  (CIN * KKT)         // 1152 contraction length

typedef __attribute__((ext_vector_type(16))) __bf16 v16bf;
typedef __attribute__((ext_vector_type(8)))  float  v8f;

// ---------------------------------------------------------------------------
// Kernel 1: pre-pack deform weights  w_def[o][c][kk] (fp32)
//           -> wpack[k][o] (bf16), k = kk*128 + c   (288 KB, stays in L2)
// ---------------------------------------------------------------------------
__global__ __launch_bounds__(256) void k_prepack(const float* __restrict__ w_def,
                                                 __bf16* __restrict__ wpack) {
    int i = blockIdx.x * 256 + threadIdx.x;     // 0 .. 1152*128-1
    int k = i >> 7;                             // contraction index
    int o = i & 127;                            // output channel
    int kk  = k >> 7;
    int cin = k & 127;
    wpack[i] = (__bf16)w_def[(o * CIN + cin) * KKT + kk];
}

// ---------------------------------------------------------------------------
// Kernel 2: offset (18ch) + mask (9ch, sigmoid) 3x3 conv, fp32.
// One thread per pixel; weights staged through LDS in 32-cin chunks
// (uniform-address broadcast reads -> conflict free). om[pixel][28].
// ---------------------------------------------------------------------------
__global__ __launch_bounds__(256) void k_offmask(const float* __restrict__ x,
                                                 const float* __restrict__ w_off,
                                                 const float* __restrict__ b_off,
                                                 const float* __restrict__ w_mask,
                                                 const float* __restrict__ b_mask,
                                                 float* __restrict__ om) {
    __shared__ float wl[27 * 32 * 9];           // 31104 B
    const int t  = threadIdx.x;
    const int gp = blockIdx.x * 256 + t;        // exact grid: 288*256 = NPIX
    const int b  = gp / HWPIX;
    const int r  = gp % HWPIX;
    const int h  = r / WWW;
    const int w  = r % WWW;

    float acc[27];
#pragma unroll
    for (int o = 0; o < 18; ++o) acc[o] = b_off[o];
#pragma unroll
    for (int o = 0; o < 9; ++o)  acc[18 + o] = b_mask[o];

    const float* xb = x + (size_t)b * CIN * HWPIX;

    for (int cb = 0; cb < CIN; cb += 32) {
        for (int i = t; i < 27 * 32 * 9; i += 256) {
            int o   = i / 288;
            int rr  = i % 288;          // cl*9 + tap
            int cl  = rr / 9;
            int tap = rr % 9;
            int cin = cb + cl;
            wl[i] = (o < 18) ? w_off[(o * CIN + cin) * 9 + tap]
                             : w_mask[((o - 18) * CIN + cin) * 9 + tap];
        }
        __syncthreads();

        for (int cl = 0; cl < 32; ++cl) {
            const float* xc = xb + (size_t)(cb + cl) * HWPIX;
#pragma unroll
            for (int ky = 0; ky < 3; ++ky) {
                int yy = h + ky - 1;
                bool vy = (yy >= 0) && (yy < HH);
#pragma unroll
                for (int kx = 0; kx < 3; ++kx) {
                    int xx = w + kx - 1;
                    bool v = vy && (xx >= 0) && (xx < WWW);
                    float xv = v ? xc[yy * WWW + xx] : 0.0f;
                    int tap = ky * 3 + kx;
#pragma unroll
                    for (int o = 0; o < 27; ++o)
                        acc[o] = fmaf(xv, wl[o * 288 + cl * 9 + tap], acc[o]);
                }
            }
        }
        __syncthreads();
    }

#pragma unroll
    for (int o = 0; o < 18; ++o) om[(size_t)gp * 28 + o] = acc[o];
#pragma unroll
    for (int o = 0; o < 9; ++o)
        om[(size_t)gp * 28 + 18 + o] = 1.0f / (1.0f + expf(-acc[18 + o]));
}

// ---------------------------------------------------------------------------
// Kernel 3: deformable gather + WMMA GEMM.
// WG = 256 thr (8 waves) -> 32 pixels (one row segment) x 128 out channels.
// Outer loop over the 9 taps (bilinear coefficients hoisted: computed once
// per tap), inner loop over 4 cin-blocks of 32 -> 36 K-steps total.
// A tile (32px x 32cin bf16) is double-buffered in LDS: one barrier per
// K-step, gather of step k+1 overlaps the WMMAs of step k.
// Each wave: tiles (mtile = wave&1, n0 = (wave>>1)*32) -> 2 WMMAs / step.
// C tiles transposed through padded LDS -> coalesced NCHW stores of y.
// ---------------------------------------------------------------------------
__global__ __launch_bounds__(256) void k_deform(const float* __restrict__ x,
                                                const float* __restrict__ om,
                                                const __bf16* __restrict__ wpack,
                                                float* __restrict__ yws) {
    __shared__ __align__(16) unsigned char smem[32 * 133 * 4];   // 17 KB
    __bf16* lA = (__bf16*)smem;     // 2 x [32 px][34] (padded bf16) in K loop
    float*  lC = (float*)smem;      // [32 px][133]   (padded f32) at writeback

    const int t    = threadIdx.x;
    const int lane = t & 31;
    const int wave = t >> 5;

    const int gp0 = blockIdx.x * 32;        // 32 pixels, same (b,h) row
    const int b   = gp0 / HWPIX;
    const int rem = gp0 % HWPIX;
    const int h   = rem / WWW;
    const int w0  = rem % WWW;

    // A-build mapping: lane -> pixel, wave id -> cin subgroup
    const int px  = lane;
    const int sub = wave;                   // 0..7, each handles 4 cin
    const int gp  = gp0 + px;
    const int xw  = w0 + px;

    // wave tile mapping
    const int mtile = wave & 1;
    const int n0    = (wave >> 1) * 32;

    // A fragment indices (per ISA 16-bit A layout), invariant over K loop
    const int apx   = mtile * 16 + (lane & 15);
    const int khalf = (lane >> 4) ? 8 : 0;

    const float* xb = x + (size_t)b * CIN * HWPIX;

    v8f c0 = {};
    v8f c1 = {};

    for (int kk = 0; kk < KKT; ++kk) {
        // --- bilinear sample coefficients for (pixel, kk): hoisted per tap ---
        const float dy = om[(size_t)gp * 28 + 2 * kk];
        const float dx = om[(size_t)gp * 28 + 2 * kk + 1];
        const float mk = om[(size_t)gp * 28 + 18 + kk];
        const int ky = kk / 3, kx = kk % 3;
        const float py  = (float)h  - 1.0f + (float)ky + dy;
        const float pxf = (float)xw - 1.0f + (float)kx + dx;
        const float fy0 = floorf(py), fx0 = floorf(pxf);
        const float wy = py - fy0,  wx = pxf - fx0;
        const int y0 = (int)fy0, x0 = (int)fx0;
        const int y1 = y0 + 1,  x1 = x0 + 1;
        const bool vy0 = (y0 >= 0) && (y0 < HH);
        const bool vy1 = (y1 >= 0) && (y1 < HH);
        const bool vx0 = (x0 >= 0) && (x0 < WWW);
        const bool vx1 = (x1 >= 0) && (x1 < WWW);
        const float omy = 1.0f - wy, omx = 1.0f - wx;
        const float w00 = omy * omx * ((vy0 && vx0) ? 1.0f : 0.0f);
        const float w01 = omy * wx  * ((vy0 && vx1) ? 1.0f : 0.0f);
        const float w10 = wy  * omx * ((vy1 && vx0) ? 1.0f : 0.0f);
        const float w11 = wy  * wx  * ((vy1 && vx1) ? 1.0f : 0.0f);
        const int cy0 = min(max(y0, 0), HH - 1);
        const int cy1 = min(max(y1, 0), HH - 1);
        const int cx0 = min(max(x0, 0), WWW - 1);
        const int cx1 = min(max(x1, 0), WWW - 1);
        const int i00 = cy0 * WWW + cx0, i01 = cy0 * WWW + cx1;
        const int i10 = cy1 * WWW + cx0, i11 = cy1 * WWW + cx1;

#pragma unroll
        for (int cb = 0; cb < 4; ++cb) {            // cin block of 32
            const int kstep = kk * 4 + cb;
            __bf16* lAb = lA + (kstep & 1) * (32 * 34);   // double buffer

            // --- build A tile: 4 channels per thread ---
#pragma unroll
            for (int i = 0; i < 4; ++i) {
                const int cl = sub * 4 + i;               // local k 0..31
                const float* xc = xb + (size_t)(cb * 32 + cl) * HWPIX;
                float v = w00 * xc[i00] + w01 * xc[i01] +
                          w10 * xc[i10] + w11 * xc[i11];
                lAb[px * 34 + cl] = (__bf16)(v * mk);
            }
            __syncthreads();   // build -> consume (dscnt drained by compiler)

            // --- load fragments (ISA 16-bit A / B layouts) and WMMA ---
            v16bf a;
#pragma unroll
            for (int j = 0; j < 16; ++j) {
                const int kl = khalf + (j & 7) + ((j & 8) ? 16 : 0);
                a[j] = lAb[apx * 34 + kl];
            }
            const __bf16* bp = wpack + ((size_t)(kstep * 32) * COUT + n0 + (lane & 15));
            v16bf bf0, bf1;
#pragma unroll
            for (int j = 0; j < 16; ++j) {
                const int kl = khalf + (j & 7) + ((j & 8) ? 16 : 0);
                bf0[j] = bp[(size_t)kl * COUT];
                bf1[j] = bp[(size_t)kl * COUT + 16];
            }

            c0 = __builtin_amdgcn_wmma_f32_16x16x32_bf16(false, a, false, bf0,
                                                         (short)0, c0, false, false);
            c1 = __builtin_amdgcn_wmma_f32_16x16x32_bf16(false, a, false, bf1,
                                                         (short)0, c1, false, false);
            // no trailing barrier: next step writes the other A buffer; this
            // buffer is only overwritten after the *next* barrier, by which
            // time every thread's ds reads here have drained.
        }
    }

    __syncthreads();   // all A-buffer traffic done before lC reuses smem

    // --- writeback: C fragments -> padded LDS -> coalesced NCHW stores ---
#pragma unroll
    for (int r = 0; r < 8; ++r) {
        const int ppx = mtile * 16 + r + ((lane >> 4) << 3);   // pixel (M)
        const int oc  = n0 + (lane & 15);                      // channel (N)
        lC[ppx * 133 + oc]      = c0[r];
        lC[ppx * 133 + oc + 16] = c1[r];
    }
    __syncthreads();
#pragma unroll
    for (int j = 0; j < 16; ++j) {
        const int e    = t + 256 * j;          // 0..4095
        const int wloc = e & 31;               // lanes -> consecutive w
        const int oc   = e >> 5;
        yws[((size_t)(b * COUT + oc)) * HWPIX + h * WWW + (w0 + wloc)] =
            lC[wloc * 133 + oc];
    }
}

// ---------------------------------------------------------------------------
// Kernel 4: per-channel mean / rstd (deterministic tree reduction).
// ---------------------------------------------------------------------------
__global__ __launch_bounds__(256) void k_stats(const float* __restrict__ yws,
                                               float* __restrict__ stats) {
    __shared__ float s1[256];
    __shared__ float s2[256];
    const int c = blockIdx.x;
    const int t = threadIdx.x;
    float sum = 0.0f, ssq = 0.0f;
    for (int b = 0; b < BB; ++b) {
        const float* p = yws + ((size_t)(b * COUT + c)) * HWPIX;
        for (int i = t; i < HWPIX; i += 256) {
            float v = p[i];
            sum += v;
            ssq += v * v;
        }
    }
    s1[t] = sum; s2[t] = ssq;
    __syncthreads();
    for (int off = 128; off > 0; off >>= 1) {
        if (t < off) { s1[t] += s1[t + off]; s2[t] += s2[t + off]; }
        __syncthreads();
    }
    if (t == 0) {
        const float inv = 1.0f / (float)NPIX;
        float m   = s1[0] * inv;
        float var = s2[0] * inv - m * m;
        stats[c]        = m;
        stats[COUT + c] = rsqrtf(var + 1e-5f);
    }
}

// ---------------------------------------------------------------------------
// Kernel 5: BN apply + ReLU, elementwise NCHW, fully coalesced.
// ---------------------------------------------------------------------------
__global__ __launch_bounds__(256) void k_bn(const float* __restrict__ yws,
                                            const float* __restrict__ stats,
                                            const float* __restrict__ gamma,
                                            const float* __restrict__ beta,
                                            float* __restrict__ out) {
    const size_t i = (size_t)blockIdx.x * 256 + threadIdx.x;
    const int c = (int)((i / HWPIX) % COUT);
    const float v = yws[i];
    const float r = (v - stats[c]) * stats[COUT + c] * gamma[c] + beta[c];
    out[i] = fmaxf(r, 0.0f);
}

// ---------------------------------------------------------------------------
// Host-side launch (stream-ordered; deterministic; no atomics, no syncs)
// ---------------------------------------------------------------------------
extern "C" void kernel_launch(void* const* d_in, const int* in_sizes, int n_in,
                              void* d_out, int out_size, void* d_ws, size_t ws_size,
                              hipStream_t stream) {
    (void)in_sizes; (void)n_in; (void)out_size; (void)ws_size;

    const float* x      = (const float*)d_in[0];
    const float* w_off  = (const float*)d_in[1];
    const float* b_off  = (const float*)d_in[2];
    const float* w_mask = (const float*)d_in[3];
    const float* b_mask = (const float*)d_in[4];
    const float* w_def  = (const float*)d_in[5];
    const float* gamma  = (const float*)d_in[6];
    const float* beta   = (const float*)d_in[7];
    float* out = (float*)d_out;

    // workspace layout (bytes)
    unsigned char* wsb = (unsigned char*)d_ws;
    const size_t OM_OFF = 0;                                 // 73728*28*4
    const size_t Y_OFF  = OM_OFF + (size_t)NPIX * 28 * 4;    // +37.7MB
    const size_t WP_OFF = Y_OFF + (size_t)NPIX * COUT * 4;   // bf16, 32B-aligned
    const size_t ST_OFF = WP_OFF + (size_t)KTOT * COUT * 2;

    float*  om    = (float*)(wsb + OM_OFF);
    float*  yws   = (float*)(wsb + Y_OFF);
    __bf16* wpack = (__bf16*)(wsb + WP_OFF);
    float*  stats = (float*)(wsb + ST_OFF);

    k_prepack<<<(KTOT * COUT) / 256, 256, 0, stream>>>(w_def, wpack);
    k_offmask<<<NPIX / 256, 256, 0, stream>>>(x, w_off, b_off, w_mask, b_mask, om);
    k_deform <<<NPIX / 32, 256, 0, stream>>>(x, om, wpack, yws);
    k_stats  <<<COUT, 256, 0, stream>>>(yws, stats);
    k_bn     <<<((size_t)NPIX * COUT) / 256, 256, 0, stream>>>(yws, stats, gamma, beta, out);
}
